// SE3CGMessageBlock_38732015075526
// MI455X (gfx1250) — matrix operand order repaired and checked by reading, not compiled, not run on hardware
//
#include <hip/hip_runtime.h>
#include <hip/hip_bf16.h>
#include <cstddef>

// Problem constants (B, N, D, R) from the reference
#define Bv 2
#define Nv 256
#define Dv 128
#define Rv 32

#define SQRT3 1.7320508075688772f
#define C15   3.872983346207417f
#define C5H   1.118033988749895f   // 0.5*sqrt(5)

typedef __attribute__((ext_vector_type(16))) _Float16 v16h;
typedef __attribute__((ext_vector_type(8)))  _Float16 v8h;
typedef __attribute__((ext_vector_type(8)))  float    v8f;

union H8 { float4 f4; v8h h8; };

__device__ inline v16h cat8(v8h lo, v8h hi) {
  v16h r;
#pragma unroll
  for (int e = 0; e < 8; ++e) { r[e] = lo[e]; r[e + 8] = hi[e]; }
  return r;
}

// fast sigmoid: v_exp + v_add + v_rcp (no IEEE divide expansion)
__device__ inline float fast_sigmoid(float x) {
  return __builtin_amdgcn_rcpf(1.0f + __expf(-x));
}

// ---------------------------------------------------------------------------
// K0: pack weights (f32 -> f16 fragment-major) and biases (f32, contiguous).
// Fragment = 512 halfs: lane (0..31) x element (0..15).
// B-matrix (32x16, f16) layout: col n = lane&15, K = (lane>>4)*8 + (e&7) + (e>>3)*16.
// w1 (32x128): 8 fragments. w2 (128x128): 4 K-tiles x 8 n-tiles = 32 frags.
// Per-radial weight stride: 20480 halfs. Bias block: p*256 + {b1:0..127, b2:128..255}.
// ---------------------------------------------------------------------------
__global__ void k0_pack(const float* __restrict__ w1_0, const float* __restrict__ w2_0,
                        const float* __restrict__ w1_1, const float* __restrict__ w2_1,
                        const float* __restrict__ w1_2, const float* __restrict__ w2_2,
                        const float* __restrict__ w1_3, const float* __restrict__ w2_3,
                        const float* __restrict__ b1_0, const float* __restrict__ b2_0,
                        const float* __restrict__ b1_1, const float* __restrict__ b2_1,
                        const float* __restrict__ b1_2, const float* __restrict__ b2_2,
                        const float* __restrict__ b1_3, const float* __restrict__ b2_3,
                        _Float16* __restrict__ wpack, float* __restrict__ biasF) {
  int idx = blockIdx.x * blockDim.x + threadIdx.x;
  if (idx >= 4 * 20480 + 1024) return;
  if (idx >= 4 * 20480) {                   // bias copy: 4 radials x 256
    int r = idx - 4 * 20480;
    int p = r >> 8;
    int q = r & 255;
    const float* b1 = (p == 0) ? b1_0 : (p == 1) ? b1_1 : (p == 2) ? b1_2 : b1_3;
    const float* b2 = (p == 0) ? b2_0 : (p == 1) ? b2_1 : (p == 2) ? b2_2 : b2_3;
    biasF[r] = (q < 128) ? b1[q] : b2[q - 128];
    return;
  }
  int p = idx / 20480;
  int rem = idx - p * 20480;
  const float* w1 = (p == 0) ? w1_0 : (p == 1) ? w1_1 : (p == 2) ? w1_2 : w1_3;
  const float* w2 = (p == 0) ? w2_0 : (p == 1) ? w2_1 : (p == 2) ? w2_2 : w2_3;
  float val;
  if (rem < 4096) {                         // w1 fragments
    int t = rem >> 9, q = rem & 511;
    int lane = q >> 4, e = q & 15;
    int n = lane & 15, kb = (lane >> 4) << 3;
    int k = kb + (e & 7) + ((e >> 3) << 4);
    val = w1[k * Dv + t * 16 + n];
  } else {                                  // w2 fragments
    rem -= 4096;
    int f = rem >> 9, q = rem & 511;
    int k2 = f >> 3, t2 = f & 7;
    int lane = q >> 4, e = q & 15;
    int n = lane & 15, kb = (lane >> 4) << 3;
    int k = k2 * 32 + kb + (e & 7) + ((e >> 3) << 4);
    val = w2[k * Dv + t2 * 16 + n];
  }
  wpack[idx] = (_Float16)val;
}

// ---------------------------------------------------------------------------
// K1: h[row, d] = node_s[row, :] @ src_w[:, d]   (row = b*N + j)
// ---------------------------------------------------------------------------
__global__ __launch_bounds__(128) void k1_h(const float* __restrict__ node_s,
                                            const float* __restrict__ src_w,
                                            float* __restrict__ hW) {
  __shared__ float srow[Dv];
  int tid = threadIdx.x;
  size_t row = blockIdx.x;
  srow[tid] = node_s[row * Dv + tid];
  __syncthreads();
  float acc = 0.f;
#pragma unroll 8
  for (int k = 0; k < Dv; ++k) acc += srow[k] * src_w[k * Dv + tid];
  hW[row * Dv + tid] = acc;
}

// ---------------------------------------------------------------------------
// K2: hot edge kernel. One workgroup (128 threads = 4 waves) per (b,i).
// Each wave owns j-tiles jt = wave, wave+4, wave+8, wave+12 (16 edges each).
// Radial loop kept ROLLED (one radial's fragments live at a time) to keep
// VGPR pressure sane; all per-radial operands are base + p*stride.
// ---------------------------------------------------------------------------
__global__ __launch_bounds__(128) void k2_edges(
    const float* __restrict__ rbf, const float* __restrict__ rhat,
    const float* __restrict__ mask, const float* __restrict__ node_v,
    const float* __restrict__ hW, const _Float16* __restrict__ wpack,
    const float* __restrict__ biasF,
    const float* __restrict__ v_scale, const float* __restrict__ t_scale,
    float* __restrict__ SW, float* __restrict__ outv, float* __restrict__ outt) {
  __shared__ float    hjS[4][16 * Dv];   // per-wave h_j tile (h[j]*mask), f32
  __shared__ _Float16 h1S[4][16 * Dv];   // per-wave layer-1 hidden, f16
  __shared__ float    rhS[4][16 * 4];    // per-wave (x,y,z,mask) per j
  __shared__ float    nvS[3 * Dv];       // node_v[b,i]
  __shared__ float    accS[10 * Dv];     // s | v0..v2 | t0..t4 | sv

  const int tid  = threadIdx.x;
  const int wave = tid >> 5;
  const int lane = tid & 31;
  const int n    = lane & 15;       // C/D col, A row, B col
  const int hi   = lane >> 4;       // half-group select
  const int m_a  = n;               // A-matrix row for this lane
  const int khalf = hi * 8;         // A-matrix K sub-offset

  const int blk = blockIdx.x;
  const int b = blk >> 8;           // N = 256
  const int i = blk & 255;
  const size_t row = (size_t)b * Nv + i;

  for (int idx = tid; idx < 10 * Dv; idx += 128) accS[idx] = 0.f;
  const size_t nvbase = row * 3 * Dv;
  for (int idx = tid; idx < 3 * Dv; idx += 128) nvS[idx] = node_v[nvbase + idx];
  __syncthreads();

  float*    hjW = &hjS[wave][0];
  _Float16* h1W = &h1S[wave][0];
  float*    rhW = &rhS[wave][0];

  for (int jt = wave; jt < 16; jt += 4) {
    const int j0 = jt * 16;

    // --- load r_hat + mask for this j-tile (lanes 0..15, one row each) ---
    if (lane < 16) {
      const size_t eb = (row * Nv + (size_t)(j0 + lane));
      const float* rp = rhat + eb * 3;
      rhW[lane * 4 + 0] = rp[0];
      rhW[lane * 4 + 1] = rp[1];
      rhW[lane * 4 + 2] = rp[2];
      rhW[lane * 4 + 3] = mask[eb];
    }
    __builtin_amdgcn_wave_barrier();

    // --- load h_j tile: h[b, j0+m, :] * mask[b,i,j0+m] ---
    for (int m = 0; m < 16; ++m) {
      float mk = rhW[m * 4 + 3];
      const float4 hv = *(const float4*)(hW + ((size_t)b * Nv + j0 + m) * Dv + lane * 4);
      float4 o;
      o.x = hv.x * mk; o.y = hv.y * mk; o.z = hv.z * mk; o.w = hv.w * mk;
      *(float4*)(hjW + m * Dv + lane * 4) = o;
    }

    // --- build rbf A-fragment (16 edges x K=32, f16 A layout) ---
    const float* rp = rbf + (row * Nv + (size_t)(j0 + m_a)) * Rv;
    float4 f0 = *(const float4*)(rp + khalf);
    float4 f1 = *(const float4*)(rp + khalf + 4);
    float4 f2 = *(const float4*)(rp + 16 + khalf);
    float4 f3 = *(const float4*)(rp + 16 + khalf + 4);
    v16h aRbf;
    aRbf[0]  = (_Float16)f0.x; aRbf[1]  = (_Float16)f0.y;
    aRbf[2]  = (_Float16)f0.z; aRbf[3]  = (_Float16)f0.w;
    aRbf[4]  = (_Float16)f1.x; aRbf[5]  = (_Float16)f1.y;
    aRbf[6]  = (_Float16)f1.z; aRbf[7]  = (_Float16)f1.w;
    aRbf[8]  = (_Float16)f2.x; aRbf[9]  = (_Float16)f2.y;
    aRbf[10] = (_Float16)f2.z; aRbf[11] = (_Float16)f2.w;
    aRbf[12] = (_Float16)f3.x; aRbf[13] = (_Float16)f3.y;
    aRbf[14] = (_Float16)f3.z; aRbf[15] = (_Float16)f3.w;

#pragma unroll 1
    for (int p = 0; p < 4; ++p) {
      const _Float16* wp1 = wpack + p * 20480;
      const _Float16* wp2 = wp1 + 4096;
      const float* b1p = biasF + p * 256;
      const float* b2p = b1p + 128;

      // ---------- layer 1: hidden = SiLU(rbf @ w1 + b1) ----------
      for (int t = 0; t < 8; ++t) {
        const _Float16* fp = wp1 + t * 512 + lane * 16;
        H8 u0, u1;
        u0.f4 = ((const float4*)fp)[0];
        u1.f4 = ((const float4*)fp)[1];
        v16h bf = cat8(u0.h8, u1.h8);
        float bias = b1p[t * 16 + n];
        v8f c;
#pragma unroll
        for (int r = 0; r < 8; ++r) c[r] = bias;
        c = __builtin_amdgcn_wmma_f32_16x16x32_f16(false, aRbf, false, bf,
                                                   (short)0, c, false, false);
#pragma unroll
        for (int r = 0; r < 8; ++r) {
          float x = c[r];
          float sl = x * fast_sigmoid(x);
          h1W[(r + 8 * hi) * Dv + t * 16 + n] = (_Float16)sl;
        }
      }
      __builtin_amdgcn_wave_barrier();

      // ---------- layer 2 + Y-weighted j-reduction ----------
      for (int t2 = 0; t2 < 8; ++t2) {
        float bias2 = b2p[t2 * 16 + n];
        v8f c2;
#pragma unroll
        for (int r = 0; r < 8; ++r) c2[r] = bias2;
#pragma unroll
        for (int k2 = 0; k2 < 4; ++k2) {
          const _Float16* ap = h1W + m_a * Dv + k2 * 32 + khalf;
          H8 a0, a1;
          a0.f4 = *(const float4*)ap;
          a1.f4 = *(const float4*)(ap + 16);
          v16h af = cat8(a0.h8, a1.h8);
          const _Float16* fp = wp2 + (k2 * 8 + t2) * 512 + lane * 16;
          H8 u0, u1;
          u0.f4 = ((const float4*)fp)[0];
          u1.f4 = ((const float4*)fp)[1];
          v16h bf = cat8(u0.h8, u1.h8);
          c2 = __builtin_amdgcn_wmma_f32_16x16x32_f16(false, af, false, bf,
                                                      (short)0, c2, false, false);
        }

        const int d_ = t2 * 16 + n;   // lane-fixed channel index
        if (p == 0) {                 // msg_s: sum_j h_j * rad0
          float s = 0.f;
#pragma unroll
          for (int r = 0; r < 8; ++r) {
            int m = r + 8 * hi;
            s += hjW[m * Dv + d_] * c2[r];
          }
          atomicAdd(&accS[d_], s);
        } else if (p == 1) {          // delta_v: sum_j (h_j*rad1) * Y1[j,c]
          float a0 = 0.f, a1 = 0.f, a2 = 0.f;
#pragma unroll
          for (int r = 0; r < 8; ++r) {
            int m = r + 8 * hi;
            float t = hjW[m * Dv + d_] * c2[r];
            a0 += SQRT3 * rhW[m * 4 + 0] * t;
            a1 += SQRT3 * rhW[m * 4 + 1] * t;
            a2 += SQRT3 * rhW[m * 4 + 2] * t;
          }
          atomicAdd(&accS[1 * Dv + d_], a0);
          atomicAdd(&accS[2 * Dv + d_], a1);
          atomicAdd(&accS[3 * Dv + d_], a2);
        } else if (p == 2) {          // delta_t: sum_j (h_j*rad2) * Y2[j,c]
          float a[5] = {0.f, 0.f, 0.f, 0.f, 0.f};
#pragma unroll
          for (int r = 0; r < 8; ++r) {
            int m = r + 8 * hi;
            float x = rhW[m * 4 + 0], y = rhW[m * 4 + 1], z = rhW[m * 4 + 2];
            float t = hjW[m * Dv + d_] * c2[r];
            a[0] += C15 * x * y * t;
            a[1] += C15 * y * z * t;
            a[2] += C5H * (3.f * z * z - 1.f) * t;
            a[3] += C15 * x * z * t;
            a[4] += 0.5f * C15 * (x * x - y * y) * t;
          }
#pragma unroll
          for (int c = 0; c < 5; ++c) atomicAdd(&accS[(4 + c) * Dv + d_], a[c]);
        } else {                      // msg_s_from_v: sum_j (v.Y1) * rad110 * mask
          float vx = nvS[d_], vy = nvS[Dv + d_], vz = nvS[2 * Dv + d_];
          float s = 0.f;
#pragma unroll
          for (int r = 0; r < 8; ++r) {
            int m = r + 8 * hi;
            float x = rhW[m * 4 + 0], y = rhW[m * 4 + 1], z = rhW[m * 4 + 2];
            float mk = rhW[m * 4 + 3];
            float dot = SQRT3 * (vx * x + vy * y + vz * z);
            s += dot * mk * c2[r];
          }
          atomicAdd(&accS[9 * Dv + d_], s);
        }
      }
    }
  }

  __syncthreads();
  // write outputs: delta_v / delta_t scaled, scalar sum -> workspace
  const int d = tid;
  SW[row * Dv + d] = accS[d] + accS[9 * Dv + d];
  float vs = v_scale[d], ts = t_scale[d];
#pragma unroll
  for (int c = 0; c < 3; ++c)
    outv[(row * 3 + c) * Dv + d] = accS[(1 + c) * Dv + d] * vs;
#pragma unroll
  for (int c = 0; c < 5; ++c)
    outt[(row * 5 + c) * Dv + d] = accS[(4 + c) * Dv + d] * ts;
}

// ---------------------------------------------------------------------------
// K3: delta_s = LayerNorm(msg_s + msg_s_from_v) @ out_w + out_b
// ---------------------------------------------------------------------------
__global__ __launch_bounds__(128) void k3_out(const float* __restrict__ SW,
                                              const float* __restrict__ ln_g,
                                              const float* __restrict__ ln_b,
                                              const float* __restrict__ out_w,
                                              const float* __restrict__ out_b,
                                              float* __restrict__ outs) {
  __shared__ float red[Dv], red2[Dv], xn[Dv];
  int tid = threadIdx.x;
  size_t row = blockIdx.x;
  float s = SW[row * Dv + tid];
  red[tid] = s; red2[tid] = s * s;
  __syncthreads();
  for (int st = 64; st > 0; st >>= 1) {
    if (tid < st) { red[tid] += red[tid + st]; red2[tid] += red2[tid + st]; }
    __syncthreads();
  }
  float m   = red[0] * (1.f / Dv);
  float var = red2[0] * (1.f / Dv) - m * m;
  float inv = rsqrtf(var + 1e-5f);
  xn[tid] = (s - m) * inv * ln_g[tid] + ln_b[tid];
  __syncthreads();
  float acc = out_b[tid];
#pragma unroll 8
  for (int k = 0; k < Dv; ++k) acc += xn[k] * out_w[k * Dv + tid];
  outs[row * Dv + tid] = acc;
}

// ---------------------------------------------------------------------------
extern "C" void kernel_launch(void* const* d_in, const int* in_sizes, int n_in,
                              void* d_out, int out_size, void* d_ws, size_t ws_size,
                              hipStream_t stream) {
  (void)in_sizes; (void)n_in; (void)out_size; (void)ws_size;
  const float* node_s = (const float*)d_in[0];
  const float* node_v = (const float*)d_in[1];
  // d_in[2] = node_t (unused by the reference)
  const float* rbf    = (const float*)d_in[3];
  const float* rhat   = (const float*)d_in[4];
  const float* maskp  = (const float*)d_in[5];
  const float* r0_w1 = (const float*)d_in[6],  *r0_b1 = (const float*)d_in[7];
  const float* r0_w2 = (const float*)d_in[8],  *r0_b2 = (const float*)d_in[9];
  const float* r1_w1 = (const float*)d_in[10], *r1_b1 = (const float*)d_in[11];
  const float* r1_w2 = (const float*)d_in[12], *r1_b2 = (const float*)d_in[13];
  const float* r2_w1 = (const float*)d_in[14], *r2_b1 = (const float*)d_in[15];
  const float* r2_w2 = (const float*)d_in[16], *r2_b2 = (const float*)d_in[17];
  const float* r3_w1 = (const float*)d_in[18], *r3_b1 = (const float*)d_in[19];
  const float* r3_w2 = (const float*)d_in[20], *r3_b2 = (const float*)d_in[21];
  const float* src_w  = (const float*)d_in[22];
  const float* ln_g   = (const float*)d_in[23];
  const float* ln_b   = (const float*)d_in[24];
  const float* out_w  = (const float*)d_in[25];
  const float* out_b  = (const float*)d_in[26];
  const float* v_scale = (const float*)d_in[27];
  const float* t_scale = (const float*)d_in[28];

  float* out  = (float*)d_out;
  float* outs = out;                         // (B,N,D)
  float* outv = out + Bv * Nv * Dv;          // (B,N,3,D)
  float* outt = out + Bv * Nv * Dv * 4;      // (B,N,5,D)

  float* wsF   = (float*)d_ws;
  float* hW    = wsF;                            // 65536 f32
  float* SW    = wsF + Bv * Nv * Dv;             // 65536 f32
  float* biasF = wsF + 2 * Bv * Nv * Dv;         // 1024 f32
  _Float16* wpack = (_Float16*)(wsF + 2 * Bv * Nv * Dv + 1024);  // 81920 f16

  hipLaunchKernelGGL(k0_pack, dim3(324), dim3(256), 0, stream,
                     r0_w1, r0_w2, r1_w1, r1_w2, r2_w1, r2_w2, r3_w1, r3_w2,
                     r0_b1, r0_b2, r1_b1, r1_b2, r2_b1, r2_b2, r3_b1, r3_b2,
                     wpack, biasF);
  hipLaunchKernelGGL(k1_h, dim3(Bv * Nv), dim3(Dv), 0, stream, node_s, src_w, hW);
  hipLaunchKernelGGL(k2_edges, dim3(Bv * Nv), dim3(128), 0, stream,
                     rbf, rhat, maskp, node_v, hW, wpack, biasF,
                     v_scale, t_scale, SW, outv, outt);
  hipLaunchKernelGGL(k3_out, dim3(Bv * Nv), dim3(Dv), 0, stream,
                     SW, ln_g, ln_b, out_w, out_b, outs);
}
